// GeometricAttentionClassifier_88880053223892
// MI455X (gfx1250) — compile-verified, matrix operand order checked by reading, not run on hardware
//
#include <hip/hip_runtime.h>
#include <hip/hip_bf16.h>
#include <math.h>

typedef __attribute__((ext_vector_type(16))) _Float16 v16h;
typedef __attribute__((ext_vector_type(8)))  float    v8f;

#define GAMMA 10.0f
#define LOG2E 1.4426950408889634f
#define NB 8
#define NN 512
#define ND 20
#define NF 16

// Raw hardware exp2 (bare v_exp_f32, flush-to-zero below 2^-126 is fine here).
__device__ __forceinline__ float hw_exp2(float x) {
  return __builtin_amdgcn_exp2f(x);
}

// Branch-free tanh: 1 - 2/(exp(2x)+1) via native exp2 + rcp.
// exp2 flush for x < -43 gives exactly -1 (correct saturation).
__device__ __forceinline__ float fast_tanh(float x) {
  const float e = hw_exp2(x * (2.0f * LOG2E));        // exp(2x)
  return 1.0f - 2.0f * __builtin_amdgcn_rcpf(e + 1.0f);
}

// Branch-free Gaussian: exp(-gamma*u^2) = exp2(-gamma*log2e * u^2).
__device__ __forceinline__ float fast_gauss(float u) {
  return hw_exp2((-GAMMA * LOG2E) * (u * u));         // arg <= 0, underflow->0 ok
}

// ---------------------------------------------------------------------------
// Phase 1: alpha[b,n,m] = sum_f w_a[f] * (tanh((rbf(d) @ W_e)[f] + b_e[f])
//                                         - tanh(feat0[f]))
// One wave per (b,n) row; each WMMA produces 16 pairs (m values).
// A (16x32 f16): rbf values, K=20 padded to 32.  B (32x16 f16): W_e padded.
// ---------------------------------------------------------------------------
__global__ __launch_bounds__(256) void alpha_wmma_kernel(
    const float* __restrict__ coords,   // (B,N,3)
    const float* __restrict__ W_e,      // (D,F)
    const float* __restrict__ b_e,      // (F)
    const float* __restrict__ w_a,      // (F)
    float* __restrict__ alpha)          // (B,N,N)
{
  const int lane  = threadIdx.x & 31;
  const int wave  = threadIdx.x >> 5;
  const int rid   = blockIdx.x * 8 + wave;      // 0 .. B*N-1
  const int b     = rid >> 9;                   // rid / 512
  const int n     = rid & (NN - 1);

  const int f     = lane & 15;                  // feature column (B/D matrices)
  const int khalf = lane >> 4;                  // which K-half this lane packs

  // ---- Build B-matrix (W_e, K padded 20->32) in the wave32 WMMA layout ----
  v16h bm;
#pragma unroll
  for (int j = 0; j < 8; ++j) {
    const int kb = (j < 4) ? (khalf * 8 + 2 * j) : (16 + khalf * 8 + 2 * (j - 4));
#pragma unroll
    for (int t = 0; t < 2; ++t) {
      const int k = kb + t;
      const float w = (k < ND) ? W_e[k * NF + f] : 0.0f;
      bm[2 * j + t] = (_Float16)w;
    }
  }

  // ---- Per-feature d=0 baseline: tanh(rbf(0) @ W_e + b_e)[f] ----
  float feat0 = b_e[f];
#pragma unroll
  for (int k = 0; k < ND; ++k) {
    const float mu = 0.1f * (float)k;
    feat0 += fast_gauss(mu) * W_e[k * NF + f];
  }
  const float base_t = fast_tanh(feat0);
  const float wa = w_a[f];
  const float be = b_e[f];

  // ---- Row coordinate (shared by the wave) ----
  const float cx = coords[(b * NN + n) * 3 + 0];
  const float cy = coords[(b * NN + n) * 3 + 1];
  const float cz = coords[(b * NN + n) * 3 + 2];

  const int pr = lane & 15;                     // A-matrix row handled by lane

  for (int m0 = 0; m0 < NN; m0 += 16) {
    const int m = m0 + pr;
    const float dx = cx - coords[(b * NN + m) * 3 + 0];
    const float dy = cy - coords[(b * NN + m) * 3 + 1];
    const float dz = cz - coords[(b * NN + m) * 3 + 2];
    const float d  = __builtin_amdgcn_sqrtf(dx * dx + dy * dy + dz * dz + 1e-12f);

    // ---- A-matrix: Gaussian RBF values in WMMA packing ----
    v16h am;
#pragma unroll
    for (int j = 0; j < 8; ++j) {
      const int kb = (j < 4) ? (khalf * 8 + 2 * j) : (16 + khalf * 8 + 2 * (j - 4));
#pragma unroll
      for (int t = 0; t < 2; ++t) {
        const int k = kb + t;
        float r = 0.0f;
        if (k < ND) r = fast_gauss(d - 0.1f * (float)k);
        am[2 * j + t] = (_Float16)r;
      }
    }

    // ---- feat = rbf @ W_e  (16 pairs x 16 features, f32 accumulate) ----
    v8f acc = {};
    acc = __builtin_amdgcn_wmma_f32_16x16x32_f16(
        /*neg_a=*/false, am, /*neg_b=*/false, bm,
        /*c_mod=*/(short)0, acc, /*reuse_a=*/false, /*reuse_b=*/false);

    // D layout: lane holds feature f for 8 pair-rows (lanes16-31: rows r+8)
    float s[8];
#pragma unroll
    for (int r = 0; r < 8; ++r)
      s[r] = wa * (fast_tanh(acc[r] + be) - base_t);

    // Sum over the 16 features (lanes of this half) — wave32 xor shuffles
#pragma unroll
    for (int off = 1; off < 16; off <<= 1) {
#pragma unroll
      for (int r = 0; r < 8; ++r)
        s[r] += __shfl_xor(s[r], off, 32);
    }

    // lane 0 -> pairs m0+0..7, lane 16 -> pairs m0+8..15
    if (f == 0) {
      float* outp = alpha + ((size_t)(b * NN + n)) * NN + m0 + khalf * 8;
#pragma unroll
      for (int r = 0; r < 8; ++r) outp[r] = s[r];
    }
  }
}

// ---------------------------------------------------------------------------
// Phase 2a: v0[b,n] = atom_emb[0]
// ---------------------------------------------------------------------------
__global__ void init_v_kernel(const float* __restrict__ atom_emb,
                              float* __restrict__ v0)
{
  const int i = blockIdx.x * blockDim.x + threadIdx.x;
  if (i < NB * NN) v0[i] = atom_emb[0];
}

// ---------------------------------------------------------------------------
// Phase 2b: v_new[b,n] = v_old[b,n] + dot(alpha[b,n,:], v_old[b,:]) / N
// One wave per row; 512-length dot + full-wave reduction.
// ---------------------------------------------------------------------------
__global__ __launch_bounds__(256) void mp_kernel(
    const float* __restrict__ alpha,
    const float* __restrict__ v_old,
    float* __restrict__ v_new)
{
  const int lane = threadIdx.x & 31;
  const int wave = threadIdx.x >> 5;
  const int rid  = blockIdx.x * 8 + wave;       // 0 .. B*N-1
  const int b    = rid >> 9;

  const float* arow = alpha + (size_t)rid * NN;
  const float* vb   = v_old + b * NN;

  float sum = 0.0f;
#pragma unroll 4
  for (int t = lane; t < NN; t += 32)
    sum += arow[t] * vb[t];

#pragma unroll
  for (int off = 1; off < 32; off <<= 1)
    sum += __shfl_xor(sum, off, 32);

  if (lane == 0)
    v_new[rid] = v_old[rid] + sum * (1.0f / (float)NN);
}

// ---------------------------------------------------------------------------
// Phase 3: out[b] = v[b,0] * Wq[0] + bq[0]
// ---------------------------------------------------------------------------
__global__ void project_kernel(const float* __restrict__ v,
                               const float* __restrict__ Wq,
                               const float* __restrict__ bq,
                               float* __restrict__ out)
{
  const int b = threadIdx.x;
  if (b < NB) out[b] = v[b * NN] * Wq[0] + bq[0];
}

// ---------------------------------------------------------------------------
extern "C" void kernel_launch(void* const* d_in, const int* in_sizes, int n_in,
                              void* d_out, int out_size, void* d_ws, size_t ws_size,
                              hipStream_t stream) {
  const float* coords   = (const float*)d_in[0];  // (8,512,3)
  const float* atom_emb = (const float*)d_in[1];  // (1,1,1)
  const float* W_e      = (const float*)d_in[2];  // (20,16)
  const float* b_e      = (const float*)d_in[3];  // (16)
  const float* w_a      = (const float*)d_in[4];  // (16)
  const float* Wq       = (const float*)d_in[5];  // (1,1)
  const float* bq       = (const float*)d_in[6];  // (1)
  float* out = (float*)d_out;                     // (8,1)

  char* ws = (char*)d_ws;
  float* alpha = (float*)ws;                                          // 8 MB
  float* v0    = (float*)(ws + (size_t)NB * NN * NN * sizeof(float)); // 16 KB
  float* v1    = v0 + NB * NN;                                        // 16 KB

  // Phase 1: alpha via WMMA.  B*N = 4096 waves, 8 waves/block.
  alpha_wmma_kernel<<<(NB * NN) / 8, 256, 0, stream>>>(coords, W_e, b_e, w_a, alpha);

  // Phase 2: init v, then two orders of message passing (ping-pong).
  init_v_kernel<<<(NB * NN + 255) / 256, 256, 0, stream>>>(atom_emb, v0);
  mp_kernel<<<(NB * NN) / 8, 256, 0, stream>>>(alpha, v0, v1);
  mp_kernel<<<(NB * NN) / 8, 256, 0, stream>>>(alpha, v1, v0);

  // Phase 3: project node 0.
  project_kernel<<<1, 32, 0, stream>>>(v0, Wq, bq, out);
}
